// GATv2_86938728006077
// MI455X (gfx1250) — compile-verified, matrix operand order checked by reading, not run on hardware
//
#include <hip/hip_runtime.h>
#include <stdint.h>

#define IN_DIM 128
#define HID 64
#define NH 4
#define FEAT 256   // NH*HID
#define OUTD 64
#define SLOPE 0.2f

typedef __attribute__((ext_vector_type(16))) __bf16 v16bf;
typedef __attribute__((ext_vector_type(8)))  __bf16 v8bf;
typedef __attribute__((ext_vector_type(8)))  float  v8f;

__device__ __forceinline__ unsigned short f32_to_bf16_bits(float f) {
  union { float f; unsigned u; } v; v.f = f;
  unsigned r = v.u + 0x7FFFu + ((v.u >> 16) & 1u);   // round-to-nearest-even
  return (unsigned short)(r >> 16);
}

__global__ void fill_f32(float* __restrict__ p, float v, int n) {
  for (int i = blockIdx.x * blockDim.x + threadIdx.x; i < n;
       i += gridDim.x * blockDim.x)
    p[i] = v;
}

// n must be a multiple of 4 (all call sites are)
__global__ void conv_bf16(const float* __restrict__ in,
                          unsigned short* __restrict__ out, int n, int do_relu) {
  int n4 = n >> 2;
  for (int i = blockIdx.x * blockDim.x + threadIdx.x; i < n4;
       i += gridDim.x * blockDim.x) {
    float4 f = ((const float4*)in)[i];
    if (do_relu) {
      f.x = fmaxf(f.x, 0.f); f.y = fmaxf(f.y, 0.f);
      f.z = fmaxf(f.z, 0.f); f.w = fmaxf(f.w, 0.f);
    }
    ushort4 o;
    o.x = f32_to_bf16_bits(f.x); o.y = f32_to_bf16_bits(f.y);
    o.z = f32_to_bf16_bits(f.z); o.w = f32_to_bf16_bits(f.w);
    ((ushort4*)out)[i] = o;
  }
}

// C[M,NC] = A[M,K](bf16) @ W[NC,K](bf16)^T + bias
// One wave computes a 16x64 tile: 4 WMMA accumulators share one A fragment.
// Requires NC % 64 == 0, K % 32 == 0.
__global__ __launch_bounds__(256)
void wmma_gemm_bias(const unsigned short* __restrict__ A,
                    const unsigned short* __restrict__ W,
                    const float* __restrict__ bias,
                    float* __restrict__ C, int M, int K, int NC) {
  int wid  = (blockIdx.x * blockDim.x + threadIdx.x) >> 5;
  int lane = threadIdx.x & 31;
  int tilesN = NC >> 6;                       // 64-wide column tiles
  int tilesM = (M + 15) >> 4;
  if (wid >= tilesM * tilesN) return;
  int tm = wid / tilesN, tn = wid % tilesN;

  int mr = tm * 16 + (lane & 15); if (mr >= M) mr = M - 1;
  int kb = (lane >> 4) * 8;                   // 0 or 8

  const unsigned short* arow = A + (size_t)mr * K;
  const unsigned short* brow = W + (size_t)(tn * 64 + (lane & 15)) * K;

  v8f acc[4] = {v8f{}, v8f{}, v8f{}, v8f{}};
  for (int k0 = 0; k0 < K; k0 += 32) {
    v8bf alo = *(const v8bf*)(arow + k0 + kb);
    v8bf ahi = *(const v8bf*)(arow + k0 + 16 + kb);
    v16bf av = __builtin_shufflevector(alo, ahi,
        0,1,2,3,4,5,6,7,8,9,10,11,12,13,14,15);
    #pragma unroll
    for (int j = 0; j < 4; j++) {
      const unsigned short* br = brow + (size_t)j * 16 * K;
      v8bf blo = *(const v8bf*)(br + k0 + kb);
      v8bf bhi = *(const v8bf*)(br + k0 + 16 + kb);
      v16bf bv = __builtin_shufflevector(blo, bhi,
          0,1,2,3,4,5,6,7,8,9,10,11,12,13,14,15);
      acc[j] = __builtin_amdgcn_wmma_f32_16x16x32_bf16(
          false, av, false, bv, (short)0, acc[j], false, false);
    }
  }

  int rowbase = tm * 16 + ((lane >> 4) << 3);
  #pragma unroll
  for (int j = 0; j < 4; j++) {
    int col = tn * 64 + j * 16 + (lane & 15);
    float bb = bias[col];
    #pragma unroll
    for (int r = 0; r < 8; r++) {
      int row = rowbase + r;
      if (row < M) C[(size_t)row * NC + col] = acc[j][r] + bb;
    }
  }
}

__device__ __forceinline__ void atomicMaxF(float* addr, float v) {
  if (v >= 0.f) atomicMax((int*)addr, __float_as_int(v));
  else          atomicMin((unsigned int*)addr, (unsigned int)__float_as_int(v));
}

__device__ __forceinline__ float lrelu(float t) { return t > 0.f ? t : SLOPE * t; }

// One wave per edge; lane L owns channels [8L, 8L+8) (head = L>>3).
// alpha[e,h] = sum_c att[h,c]*leaky_relu(h[dst]+h[src]); segment-max into mx.
__global__ __launch_bounds__(256)
void edge_alpha_wave(const float* __restrict__ h,
                     const long long* __restrict__ ei, int E,
                     const float* __restrict__ att,
                     float* __restrict__ alpha, float* __restrict__ mx) {
  int wid  = (blockIdx.x * blockDim.x + threadIdx.x) >> 5;
  int lane = threadIdx.x & 31;
  if (wid >= E) return;
  int s = (int)ei[wid], d = (int)ei[E + wid];
  const float4* pj = (const float4*)(h + (size_t)s * FEAT) + lane * 2;
  const float4* pi = (const float4*)(h + (size_t)d * FEAT) + lane * 2;
  const float4* pa = (const float4*)att + lane * 2;
  float sum = 0.f;
  #pragma unroll
  for (int q = 0; q < 2; q++) {
    float4 a = pi[q], b = pj[q], w = pa[q];
    sum += lrelu(a.x + b.x) * w.x + lrelu(a.y + b.y) * w.y +
           lrelu(a.z + b.z) * w.z + lrelu(a.w + b.w) * w.w;
  }
  // reduce over the 8-lane group that shares a head
  sum += __shfl_xor(sum, 1, 32);
  sum += __shfl_xor(sum, 2, 32);
  sum += __shfl_xor(sum, 4, 32);
  if ((lane & 7) == 0) {
    int hd = lane >> 3;
    alpha[(size_t)wid * NH + hd] = sum;
    atomicMaxF(&mx[(size_t)d * NH + hd], sum);
  }
}

// alpha <- exp(alpha - mx[dst]); den[dst] += alpha   (thread per edge*head)
__global__ void edge_exp(const long long* __restrict__ ei, int E,
                         const float* __restrict__ mx,
                         float* __restrict__ alpha, float* __restrict__ den) {
  int idx = blockIdx.x * blockDim.x + threadIdx.x;
  if (idx >= E * NH) return;
  int e = idx >> 2, hd = idx & 3;
  int d = (int)ei[E + e];
  float ev = __expf(alpha[idx] - mx[(size_t)d * NH + hd]);
  alpha[idx] = ev;
  atomicAdd(&den[(size_t)d * NH + hd], ev);
}

// One wave per edge: out[dst] += h[src] * (alpha/den[dst]), coalesced atomics.
__global__ __launch_bounds__(256)
void edge_agg_wave(const float* __restrict__ h,
                   const long long* __restrict__ ei, int E,
                   const float* __restrict__ alpha,
                   const float* __restrict__ den,
                   float* __restrict__ out) {
  int wid  = (blockIdx.x * blockDim.x + threadIdx.x) >> 5;
  int lane = threadIdx.x & 31;
  if (wid >= E) return;
  int s = (int)ei[wid], d = (int)ei[E + wid];
  int hd = lane >> 3;
  float a = alpha[(size_t)wid * NH + hd] / den[(size_t)d * NH + hd];
  const float4* pj = (const float4*)(h + (size_t)s * FEAT) + lane * 2;
  float* po = out + (size_t)d * FEAT + lane * 8;
  #pragma unroll
  for (int q = 0; q < 2; q++) {
    float4 xj = pj[q];
    atomicAdd(po + 4 * q + 0, xj.x * a);
    atomicAdd(po + 4 * q + 1, xj.y * a);
    atomicAdd(po + 4 * q + 2, xj.z * a);
    atomicAdd(po + 4 * q + 3, xj.w * a);
  }
}

static inline int cdiv(long long a, long long b) { return (int)((a + b - 1) / b); }

extern "C" void kernel_launch(void* const* d_in, const int* in_sizes, int n_in,
                              void* d_out, int out_size, void* d_ws, size_t ws_size,
                              hipStream_t stream) {
  const float*     x    = (const float*)d_in[0];
  const long long* ei   = (const long long*)d_in[1];   // int64 [2,E]
  const float*     W1   = (const float*)d_in[2];
  const float*     b1   = (const float*)d_in[3];
  const float*     att1 = (const float*)d_in[4];
  const float*     W2   = (const float*)d_in[5];
  const float*     b2   = (const float*)d_in[6];
  const float*     att2 = (const float*)d_in[7];
  const float*     W3   = (const float*)d_in[8];
  const float*     b3   = (const float*)d_in[9];
  const float*     W4   = (const float*)d_in[10];
  const float*     b4   = (const float*)d_in[11];

  const int N = in_sizes[0] / IN_DIM;
  const int E = in_sizes[1] / 2;

  // workspace carve-up (all 16B-aligned segments)
  float*          h0   = (float*)d_ws;                                  // N*FEAT f32
  float*          agg  = h0 + (size_t)N * FEAT;                         // N*FEAT f32
  unsigned short* fb   = (unsigned short*)(agg + (size_t)N * FEAT);     // N*FEAT bf16
  float*          alp  = (float*)(fb + (size_t)N * FEAT);               // E*NH f32
  float*          mx   = alp + (size_t)E * NH;                          // N*NH
  float*          den  = mx + (size_t)N * NH;                           // N*NH
  unsigned short* w1b  = (unsigned short*)(den + (size_t)N * NH);
  unsigned short* w2b  = w1b + FEAT * IN_DIM;
  unsigned short* w3b  = w2b + FEAT * FEAT;
  unsigned short* w4b  = w3b + OUTD * FEAT;

  const int T = 256;
  const int EH = E * NH;
  const int edgeBlocks = cdiv((long long)E * 32, T);   // wave-per-edge kernels

  // ---- convert weights + input to bf16 ----
  conv_bf16<<<cdiv(FEAT * IN_DIM / 4, T), T, 0, stream>>>(W1, w1b, FEAT * IN_DIM, 0);
  conv_bf16<<<cdiv(FEAT * FEAT / 4, T), T, 0, stream>>>(W2, w2b, FEAT * FEAT, 0);
  conv_bf16<<<cdiv(OUTD * FEAT / 4, T), T, 0, stream>>>(W3, w3b, OUTD * FEAT, 0);
  conv_bf16<<<cdiv(OUTD * OUTD / 4, T), T, 0, stream>>>(W4, w4b, OUTD * OUTD, 0);
  conv_bf16<<<cdiv((long long)N * IN_DIM / 4, T), T, 0, stream>>>(x, fb, N * IN_DIM, 0);

  // ---- layer 1: projection (WMMA, 16x64 tiles) ----
  int tiles = cdiv(N, 16) * (FEAT / 64);
  wmma_gemm_bias<<<cdiv(tiles, 8), 256, 0, stream>>>(fb, w1b, b1, h0, N, IN_DIM, FEAT);

  // ---- layer 1: edge softmax + aggregate ----
  fill_f32<<<cdiv(N * NH, T), T, 0, stream>>>(mx, -3.4028235e38f, N * NH);
  fill_f32<<<cdiv(N * NH, T), T, 0, stream>>>(den, 0.f, N * NH);
  fill_f32<<<cdiv((long long)N * FEAT, T), T, 0, stream>>>(agg, 0.f, N * FEAT);
  edge_alpha_wave<<<edgeBlocks, T, 0, stream>>>(h0, ei, E, att1, alp, mx);
  edge_exp<<<cdiv(EH, T), T, 0, stream>>>(ei, E, mx, alp, den);
  edge_agg_wave<<<edgeBlocks, T, 0, stream>>>(h0, ei, E, alp, den, agg);

  // relu + bf16 for next GEMM
  conv_bf16<<<cdiv((long long)N * FEAT / 4, T), T, 0, stream>>>(agg, fb, N * FEAT, 1);

  // ---- layer 2: projection (WMMA, K=256) ----
  wmma_gemm_bias<<<cdiv(tiles, 8), 256, 0, stream>>>(fb, w2b, b2, h0, N, FEAT, FEAT);

  // ---- layer 2: edge softmax + aggregate ----
  fill_f32<<<cdiv(N * NH, T), T, 0, stream>>>(mx, -3.4028235e38f, N * NH);
  fill_f32<<<cdiv(N * NH, T), T, 0, stream>>>(den, 0.f, N * NH);
  fill_f32<<<cdiv((long long)N * FEAT, T), T, 0, stream>>>(agg, 0.f, N * FEAT);
  edge_alpha_wave<<<edgeBlocks, T, 0, stream>>>(h0, ei, E, att2, alp, mx);
  edge_exp<<<cdiv(EH, T), T, 0, stream>>>(ei, E, mx, alp, den);
  edge_agg_wave<<<edgeBlocks, T, 0, stream>>>(h0, ei, E, alp, den, agg);

  conv_bf16<<<cdiv((long long)N * FEAT / 4, T), T, 0, stream>>>(agg, fb, N * FEAT, 1);

  // ---- post-MP: h@W3^T + b3 (256 -> 64), then @W4^T + b4 (64 -> 64) ----
  int tiles3 = cdiv(N, 16) * (OUTD / 64);
  wmma_gemm_bias<<<cdiv(tiles3, 8), 256, 0, stream>>>(fb, w3b, b3, h0, N, FEAT, OUTD);
  conv_bf16<<<cdiv((long long)N * OUTD / 4, T), T, 0, stream>>>(h0, fb, N * OUTD, 0);
  wmma_gemm_bias<<<cdiv(tiles3, 8), 256, 0, stream>>>(fb, w4b, b4, (float*)d_out,
                                                      N, OUTD, OUTD);
}
